// Attention_32701880992003
// MI455X (gfx1250) — matrix operand order repaired and checked by reading, not compiled, not run on hardware
//
#include <hip/hip_runtime.h>
#include <hip/hip_bf16.h>
#include <math.h>

// ---------------------------------------------------------------------------
// Attention: out = softmax((xWq)(xWk)^T * scale) (xWv) Wo^T + bo
// B=4, N=2048, C=1024, H=16, D=64.  Mixed precision: f16 WMMA, f32 accum.
// Ping-pong (unroll-2) fragment buffers: no register-rotation movs.
// ---------------------------------------------------------------------------

typedef _Float16 f16;
typedef __attribute__((ext_vector_type(16))) _Float16 v16h;
typedef __attribute__((ext_vector_type(8)))  _Float16 v8h;
typedef __attribute__((ext_vector_type(8)))  float    v8f;

#define DIM 1024
#define SEQ 2048
#define NB  4
#define NH  16
#define HD  64
#define SCALE 0.125f   // 64^-0.5

// ---- WMMA wrapper: D = A(16x32 f16) * B(32x16 f16) + C(16x16 f32) ----------
__device__ __forceinline__ v8f wmma_f16(v16h a, v16h b, v8f c) {
  return __builtin_amdgcn_wmma_f32_16x16x32_f16(
      /*neg_a=*/false, a, /*neg_b=*/false, b,
      /*c_mod=*/(short)0, c, /*reuse_a=*/false, /*reuse_b=*/false);
}

// A-fragment (16x32, f16) from row-major [M,K] tile start, leading dim ld.
__device__ __forceinline__ v16h load_a_frag(const f16* __restrict__ base,
                                            int ld, int lane) {
  const int m = lane & 15, g = lane >> 4;
  const f16* p = base + (size_t)m * ld + g * 8;
  union { v16h v; v8h h2[2]; } u;
  u.h2[0] = *(const v8h*)(p);
  u.h2[1] = *(const v8h*)(p + 16);
  return u.v;
}

// B-fragment (32x16, f16) from a row-major [N,K] array.
__device__ __forceinline__ v16h load_bt_frag(const f16* __restrict__ base,
                                             int ld, int lane) {
  const int n = lane & 15, g = lane >> 4;
  const f16* p = base + (size_t)n * ld + g * 8;
  union { v16h v; v8h h2[2]; } u;
  u.h2[0] = *(const v8h*)(p);
  u.h2[1] = *(const v8h*)(p + 16);
  return u.v;
}

// Row-wise reductions over the 16 lanes holding one C-row's columns.
__device__ __forceinline__ float hmax16(float v) {
  v = fmaxf(v, __shfl_xor(v, 1));
  v = fmaxf(v, __shfl_xor(v, 2));
  v = fmaxf(v, __shfl_xor(v, 4));
  v = fmaxf(v, __shfl_xor(v, 8));
  return v;
}
__device__ __forceinline__ float hsum16(float v) {
  v += __shfl_xor(v, 1);
  v += __shfl_xor(v, 2);
  v += __shfl_xor(v, 4);
  v += __shfl_xor(v, 8);
  return v;
}

// ---------------------------------------------------------------------------
// Kernel 0: f32 -> f16 convert, 8 elements/thread
// ---------------------------------------------------------------------------
__global__ void __launch_bounds__(256) k_cvt(const float* __restrict__ in,
                                             f16* __restrict__ out, int n8) {
  int i = blockIdx.x * 256 + threadIdx.x;
  if (i < n8) {
    const float4* p = (const float4*)in + (size_t)i * 2;
    float4 x0 = p[0], x1 = p[1];
    union { v8h v; f16 e[8]; } u;
    u.e[0] = (f16)x0.x; u.e[1] = (f16)x0.y; u.e[2] = (f16)x0.z; u.e[3] = (f16)x0.w;
    u.e[4] = (f16)x1.x; u.e[5] = (f16)x1.y; u.e[6] = (f16)x1.z; u.e[7] = (f16)x1.w;
    ((v8h*)out)[i] = u.v;
  }
}

// ---------------------------------------------------------------------------
// Kernel 1: projection GEMM.  y[m, o] = sum_c A[m,c] * W[o,c]
// Block = 256 thr = 8 waves; wave tile 32 x 64; ping-pong K pipeline.
// mode 0: out[b,h,n,d]   (Q, K)      mode 1: out[b,h,d,n]   (V transposed)
// ---------------------------------------------------------------------------
__global__ void __launch_bounds__(256)
k_proj(const f16* __restrict__ A, const f16* __restrict__ W,
       f16* __restrict__ out, int mode) {
  const int lane = threadIdx.x & 31;
  const int w    = threadIdx.x >> 5;
  const int m0   = blockIdx.x * 256 + w * 32;
  const int n0   = blockIdx.y * 64;

  const f16* Ar0 = A + (size_t)m0 * DIM;
  const f16* Ar1 = A + (size_t)(m0 + 16) * DIM;
  const f16* Wr  = W + (size_t)n0 * DIM;

  v8f acc[2][4] = {};

  // buffer A <- k = 0
  v16h a0A = load_a_frag(Ar0, DIM, lane);
  v16h a1A = load_a_frag(Ar1, DIM, lane);
  v16h bA[4];
#pragma unroll
  for (int j = 0; j < 4; ++j) bA[j] = load_bt_frag(Wr + (size_t)(16 * j) * DIM, DIM, lane);

  for (int k = 0; k < DIM; k += 64) {
    const int k1 = k + 32;
    // prefetch buffer B <- k+32
    v16h a0B = load_a_frag(Ar0 + k1, DIM, lane);
    v16h a1B = load_a_frag(Ar1 + k1, DIM, lane);
    v16h bB[4];
#pragma unroll
    for (int j = 0; j < 4; ++j)
      bB[j] = load_bt_frag(Wr + (size_t)(16 * j) * DIM + k1, DIM, lane);

    // compute with buffer A (k)
#pragma unroll
    for (int j = 0; j < 4; ++j) {
      acc[0][j] = wmma_f16(a0A, bA[j], acc[0][j]);
      acc[1][j] = wmma_f16(a1A, bA[j], acc[1][j]);
    }

    // prefetch buffer A <- k+64 (wraps to 0 on last iteration; harmless)
    const int k2 = (k + 64) & (DIM - 1);
    a0A = load_a_frag(Ar0 + k2, DIM, lane);
    a1A = load_a_frag(Ar1 + k2, DIM, lane);
#pragma unroll
    for (int j = 0; j < 4; ++j)
      bA[j] = load_bt_frag(Wr + (size_t)(16 * j) * DIM + k2, DIM, lane);

    // compute with buffer B (k+32)
#pragma unroll
    for (int j = 0; j < 4; ++j) {
      acc[0][j] = wmma_f16(a0B, bB[j], acc[0][j]);
      acc[1][j] = wmma_f16(a1B, bB[j], acc[1][j]);
    }
  }

  const int hi = lane >> 4, nl = lane & 15;
#pragma unroll
  for (int i = 0; i < 2; ++i) {
#pragma unroll
    for (int j = 0; j < 4; ++j) {
      const int col = n0 + 16 * j + nl;
      const int h = col >> 6, d = col & 63;
#pragma unroll
      for (int r = 0; r < 8; ++r) {
        const int row = m0 + 16 * i + r + 8 * hi;   // flat row in [B*N]
        const int b = row >> 11, ns = row & (SEQ - 1);
        const f16 val = (f16)acc[i][j][r];
        if (mode == 0)
          out[(((size_t)b * NH + h) * SEQ + ns) * HD + d] = val;
        else
          out[(((size_t)b * NH + h) * HD + d) * SEQ + ns] = val;
      }
    }
  }
}

// ---------------------------------------------------------------------------
// Kernel 2: flash attention.  Grid (N/64, H, B), block 128 = 4 waves.
// Each wave owns 16 query rows; kv chunks of 32; ping-pong K fragments.
// ---------------------------------------------------------------------------
__global__ void __launch_bounds__(128)
k_attn(const f16* __restrict__ Q, const f16* __restrict__ K,
       const f16* __restrict__ Vt, f16* __restrict__ ctx) {
  __shared__ __align__(32) f16 plds[4][16][32];

  const int lane = threadIdx.x & 31;
  const int w    = threadIdx.x >> 5;
  const int nl = lane & 15, hi = lane >> 4;
  const int h = blockIdx.y, b = blockIdx.z;
  const int qbase = blockIdx.x * 64 + w * 16;

  const size_t head = ((size_t)b * NH + h) * SEQ * HD;  // same extent Q/K/Vt

  const v16h a0 = load_a_frag(Q + head + (size_t)qbase * HD + 0,  HD, lane);
  const v16h a1 = load_a_frag(Q + head + (size_t)qbase * HD + 32, HD, lane);

  v8f o[4] = {};
  float mrow[8], lrow[8];
#pragma unroll
  for (int r = 0; r < 8; ++r) { mrow[r] = -1e30f; lrow[r] = 0.f; }

  // ping-pong K-fragment buffers; preload chunk 0 into buffer A
  v16h kbA[4], kbB[4];
  kbA[0] = load_bt_frag(K + head + (size_t)0 * HD + 0,   HD, lane);
  kbA[1] = load_bt_frag(K + head + (size_t)0 * HD + 32,  HD, lane);
  kbA[2] = load_bt_frag(K + head + (size_t)16 * HD + 0,  HD, lane);
  kbA[3] = load_bt_frag(K + head + (size_t)16 * HD + 32, HD, lane);

  auto chunk = [&](int c, v16h* kbc, v16h* kbn) {
    const int kv  = c * 32;
    const int kvn = ((c + 1) & (SEQ / 32 - 1)) * 32;   // wraps on last iter

    // ---- scores S(16x32) = Q(16x64) @ K_chunk^T ----
    v8f s0 = {}, s1 = {};
    s0 = wmma_f16(a0, kbc[0], s0); s0 = wmma_f16(a1, kbc[1], s0);
    s1 = wmma_f16(a0, kbc[2], s1); s1 = wmma_f16(a1, kbc[3], s1);

    // ---- issue V loads (this chunk) + K loads (next chunk) early ----
    v16h vb[4];
#pragma unroll
    for (int j = 0; j < 4; ++j)
      vb[j] = load_bt_frag(Vt + head + (size_t)(16 * j) * SEQ + kv, SEQ, lane);
    kbn[0] = load_bt_frag(K + head + (size_t)(kvn +  0) * HD + 0,  HD, lane);
    kbn[1] = load_bt_frag(K + head + (size_t)(kvn +  0) * HD + 32, HD, lane);
    kbn[2] = load_bt_frag(K + head + (size_t)(kvn + 16) * HD + 0,  HD, lane);
    kbn[3] = load_bt_frag(K + head + (size_t)(kvn + 16) * HD + 32, HD, lane);

    // ---- online softmax update (overlaps the loads above) ----
#pragma unroll
    for (int r = 0; r < 8; ++r) {
      const float x0 = s0[r] * SCALE, x1 = s1[r] * SCALE;
      const float mc = hmax16(fmaxf(x0, x1));
      const float mn = fmaxf(mrow[r], mc);
      const float alpha = __expf(mrow[r] - mn);
      const float p0 = __expf(x0 - mn), p1 = __expf(x1 - mn);
      lrow[r] = lrow[r] * alpha + hsum16(p0 + p1);
      mrow[r] = mn;
#pragma unroll
      for (int j = 0; j < 4; ++j) o[j][r] *= alpha;
      plds[w][r + 8 * hi][nl]      = (f16)p0;
      plds[w][r + 8 * hi][16 + nl] = (f16)p1;
    }
    __syncthreads();

    // ---- reload P as A-fragment (16x32) ----
    v16h pa;
    {
      union { v16h v; v8h h2[2]; } u;
      u.h2[0] = *(const v8h*)&plds[w][nl][hi * 8];
      u.h2[1] = *(const v8h*)&plds[w][nl][16 + hi * 8];
      pa = u.v;
    }
    __syncthreads();

    // ---- O(16x64) += P(16x32) @ V_chunk(32x64) ----
#pragma unroll
    for (int j = 0; j < 4; ++j) o[j] = wmma_f16(pa, vb[j], o[j]);
  };

  for (int c = 0; c < SEQ / 32; c += 2) {
    chunk(c,     kbA, kbB);   // even chunk computes from A, fills B
    chunk(c + 1, kbB, kbA);   // odd chunk computes from B, fills A
  }

  // ---- normalize and write ctx[b, n, h*64 + d] ----
#pragma unroll
  for (int j = 0; j < 4; ++j) {
    const int d = 16 * j + nl;
#pragma unroll
    for (int r = 0; r < 8; ++r) {
      const int ns = qbase + r + 8 * hi;
      const float val = o[j][r] / lrow[r];
      ctx[((size_t)b * SEQ + ns) * DIM + h * HD + d] = (f16)val;
    }
  }
}

// ---------------------------------------------------------------------------
// Kernel 3: output projection + bias, f32 result; same ping-pong pipeline.
// ---------------------------------------------------------------------------
__global__ void __launch_bounds__(256)
k_outproj(const f16* __restrict__ A, const f16* __restrict__ W,
          const float* __restrict__ bias, float* __restrict__ out) {
  const int lane = threadIdx.x & 31;
  const int w    = threadIdx.x >> 5;
  const int m0   = blockIdx.x * 256 + w * 32;
  const int n0   = blockIdx.y * 64;

  const f16* Ar0 = A + (size_t)m0 * DIM;
  const f16* Ar1 = A + (size_t)(m0 + 16) * DIM;
  const f16* Wr  = W + (size_t)n0 * DIM;

  v8f acc[2][4] = {};

  v16h a0A = load_a_frag(Ar0, DIM, lane);
  v16h a1A = load_a_frag(Ar1, DIM, lane);
  v16h bA[4];
#pragma unroll
  for (int j = 0; j < 4; ++j) bA[j] = load_bt_frag(Wr + (size_t)(16 * j) * DIM, DIM, lane);

  for (int k = 0; k < DIM; k += 64) {
    const int k1 = k + 32;
    v16h a0B = load_a_frag(Ar0 + k1, DIM, lane);
    v16h a1B = load_a_frag(Ar1 + k1, DIM, lane);
    v16h bB[4];
#pragma unroll
    for (int j = 0; j < 4; ++j)
      bB[j] = load_bt_frag(Wr + (size_t)(16 * j) * DIM + k1, DIM, lane);

#pragma unroll
    for (int j = 0; j < 4; ++j) {
      acc[0][j] = wmma_f16(a0A, bA[j], acc[0][j]);
      acc[1][j] = wmma_f16(a1A, bA[j], acc[1][j]);
    }

    const int k2 = (k + 64) & (DIM - 1);
    a0A = load_a_frag(Ar0 + k2, DIM, lane);
    a1A = load_a_frag(Ar1 + k2, DIM, lane);
#pragma unroll
    for (int j = 0; j < 4; ++j)
      bA[j] = load_bt_frag(Wr + (size_t)(16 * j) * DIM + k2, DIM, lane);

#pragma unroll
    for (int j = 0; j < 4; ++j) {
      acc[0][j] = wmma_f16(a0B, bB[j], acc[0][j]);
      acc[1][j] = wmma_f16(a1B, bB[j], acc[1][j]);
    }
  }

  const int hi = lane >> 4, nl = lane & 15;
#pragma unroll
  for (int i = 0; i < 2; ++i) {
#pragma unroll
    for (int j = 0; j < 4; ++j) {
      const int col = n0 + 16 * j + nl;
      const float bj = bias[col];
#pragma unroll
      for (int r = 0; r < 8; ++r) {
        const int row = m0 + 16 * i + r + 8 * hi;
        out[(size_t)row * DIM + col] = acc[i][j][r] + bj;
      }
    }
  }
}

// ---------------------------------------------------------------------------
// Launcher
// ---------------------------------------------------------------------------
extern "C" void kernel_launch(void* const* d_in, const int* in_sizes, int n_in,
                              void* d_out, int out_size, void* d_ws, size_t ws_size,
                              hipStream_t stream) {
  (void)in_sizes; (void)n_in; (void)out_size; (void)ws_size;

  const float* x  = (const float*)d_in[0];
  const float* wq = (const float*)d_in[1];
  const float* wk = (const float*)d_in[2];
  const float* wv = (const float*)d_in[3];
  const float* wo = (const float*)d_in[4];
  const float* bo = (const float*)d_in[5];
  float* out = (float*)d_out;

  const size_t XN = (size_t)NB * SEQ * DIM;   // 8,388,608
  const size_t WN = (size_t)DIM * DIM;        // 1,048,576

  f16* ws  = (f16*)d_ws;
  f16* xh  = ws;                 // [B*N, C]      f16
  f16* qh  = xh  + XN;           // [B,H,N,64]    f16
  f16* kh  = qh  + XN;           // [B,H,N,64]    f16
  f16* vth = kh  + XN;           // [B,H,64,N]    f16
  f16* ctx = vth + XN;           // [B,N,C]       f16
  f16* wqh = ctx + XN;
  f16* wkh = wqh + WN;
  f16* wvh = wkh + WN;
  f16* woh = wvh + WN;           // total 46.1M halves = ~88 MB

  const int xn8 = (int)(XN / 8), wn8 = (int)(WN / 8);
  k_cvt<<<dim3((xn8 + 255) / 256), 256, 0, stream>>>(x,  xh,  xn8);
  k_cvt<<<dim3((wn8 + 255) / 256), 256, 0, stream>>>(wq, wqh, wn8);
  k_cvt<<<dim3((wn8 + 255) / 256), 256, 0, stream>>>(wk, wkh, wn8);
  k_cvt<<<dim3((wn8 + 255) / 256), 256, 0, stream>>>(wv, wvh, wn8);
  k_cvt<<<dim3((wn8 + 255) / 256), 256, 0, stream>>>(wo, woh, wn8);

  dim3 ggrid(32, 16);     // M/256 x N/64
  k_proj<<<ggrid, 256, 0, stream>>>(xh, wqh, qh,  0);
  k_proj<<<ggrid, 256, 0, stream>>>(xh, wkh, kh,  0);
  k_proj<<<ggrid, 256, 0, stream>>>(xh, wvh, vth, 1);

  k_attn<<<dim3(SEQ / 64, NH, NB), 128, 0, stream>>>(qh, kh, vth, ctx);

  k_outproj<<<ggrid, 256, 0, stream>>>(ctx, woh, bo, out);
}
